// RNNModel_19980187861358
// MI455X (gfx1250) — matrix operand order repaired and checked by reading, not compiled, not run on hardware
//
#include <hip/hip_runtime.h>
#include <hip/hip_bf16.h>
#include <cstdint>

// ---------------------------------------------------------------------------
// RNN scan on gfx1250:  h_{t+1} = tanh([x_t | h_t] @ [W_ih | W_hh]^T + b)
// Persistent kernel, 16 WGs (each owns 32 of 512 hidden cols), weight slice
// resident in LDS, f16 WMMA w/ f32 accumulation, per-step device barrier,
// ping-pong h buffers, async double-buffered LDS staging of x_t.
// ---------------------------------------------------------------------------

typedef __attribute__((ext_vector_type(16))) _Float16 v16h;
typedef __attribute__((ext_vector_type(8)))  _Float16 v8h;
typedef __attribute__((ext_vector_type(8)))  float    v8f;
typedef __attribute__((ext_vector_type(4)))  int      i32x4;

// address-space-qualified pointee types for the async-to-LDS builtin
typedef __attribute__((address_space(1))) i32x4 gi32x4;  // global
typedef __attribute__((address_space(3))) i32x4 li32x4;  // LDS

static constexpr int Bz = 64;    // batch
static constexpr int Tz = 1024;  // timesteps
static constexpr int Iz = 128;   // input dim
static constexpr int Hz = 512;   // hidden dim
static constexpr int KTOT = Iz + Hz;   // 640 concatenated K
static constexpr int NWG  = 16;        // persistent workgroups
static constexpr int NCOL = Hz / NWG;  // 32 hidden cols per WG

// ---------------------------------------------------------------------------
// f32 -> f16 conversion (grid-stride)
__global__ void cvt_f32_to_f16(const float* __restrict__ in,
                               _Float16* __restrict__ out, int n) {
  int i = blockIdx.x * blockDim.x + threadIdx.x;
  int stride = gridDim.x * blockDim.x;
  for (; i < n; i += stride) out[i] = (_Float16)in[i];
}

// init: h0 = 0, bias_sum = b_ih + b_hh, barrier counter = 0
__global__ void init_state(const float* __restrict__ b_ih,
                           const float* __restrict__ b_hh,
                           float* __restrict__ bias_sum,
                           _Float16* __restrict__ h0,
                           int* __restrict__ counter) {
  int i = blockIdx.x * blockDim.x + threadIdx.x;
  if (i < Hz) bias_sum[i] = b_ih[i] + b_hh[i];
  if (i < Bz * Hz) h0[i] = (_Float16)0.0f;
  if (i == 0) *counter = 0;
}

// ---------------------------------------------------------------------------
// branch-free tanh: (e^{2x}-1)/(e^{2x}+1) via v_exp_f32 + v_rcp_f32.
// Clamp keeps exp2 finite (exp2(2*log2e*9) ~ 6.6e7); result saturates to +-1
// well inside f16 precision.
__device__ __forceinline__ float fast_tanh(float x) {
  float xc = fminf(fmaxf(x, -9.0f), 9.0f);
#if __has_builtin(__builtin_amdgcn_exp2f)
  float e = __builtin_amdgcn_exp2f(xc * 2.8853900817779268f); // 2*log2(e)
#else
  float e = exp2f(xc * 2.8853900817779268f);
#endif
#if __has_builtin(__builtin_amdgcn_rcpf)
  float r = __builtin_amdgcn_rcpf(e + 1.0f);
#else
  float r = 1.0f / (e + 1.0f);
#endif
  return (e - 1.0f) * r;
}

// ---------------------------------------------------------------------------
// async global -> LDS 16B copy (ASYNCcnt path); sync fallback keeps compile safe
__device__ __forceinline__ void async_copy_b128(const _Float16* gsrc,
                                                _Float16* ldst) {
#if __has_builtin(__builtin_amdgcn_global_load_async_to_lds_b128)
  __builtin_amdgcn_global_load_async_to_lds_b128(
      (gi32x4*)(uintptr_t)gsrc,   // inttoptr -> AS1 int4*
      (li32x4*)ldst,              // addrspacecast generic -> AS3 int4*
      0, 0);
#else
  *(v8h*)ldst = *(const v8h*)gsrc;
#endif
}

__device__ __forceinline__ void wait_async() {
#if __has_builtin(__builtin_amdgcn_s_wait_asynccnt)
  __builtin_amdgcn_s_wait_asynccnt(0);
#else
  asm volatile("s_wait_asynccnt 0x0" ::: "memory");
#endif
}

// stage x_t tile (64 rows x 128 f16 = 16 KB) into LDS: 1024 16B chunks / 256 thr
__device__ __forceinline__ void stage_x(_Float16* __restrict__ dst,
                                        const _Float16* __restrict__ x16,
                                        int t, int tid) {
#pragma unroll
  for (int j = 0; j < 4; ++j) {
    int id  = tid + 256 * j;
    int row = id >> 4;          // 0..63
    int c   = (id & 15) * 8;    // f16 offset within row
    async_copy_b128(x16 + ((size_t)row * Tz + t) * Iz + c, dst + row * Iz + c);
  }
}

// device-scope barrier across the persistent grid (monotonic counter)
__device__ __forceinline__ void grid_barrier(int* counter, int target) {
  __threadfence();
  __syncthreads();
  if (threadIdx.x == 0) {
    __hip_atomic_fetch_add(counter, 1, __ATOMIC_RELEASE, __HIP_MEMORY_SCOPE_AGENT);
    while (__hip_atomic_load(counter, __ATOMIC_ACQUIRE, __HIP_MEMORY_SCOPE_AGENT) < target) {
      __builtin_amdgcn_s_sleep(1);
    }
  }
  __syncthreads();
}

// build A fragment from one contiguous 32B chunk (k = k0 + 16*half + s, matches B)
__device__ __forceinline__ v16h a_frag(const _Float16* base, int k0, int half) {
  return *(const v16h*)(base + k0 + half * 16);
}

// ---------------------------------------------------------------------------
// Persistent recurrence kernel. Grid = 16 blocks x 256 threads (8 wave32).
__global__ void rnn_scan_kernel(const _Float16* __restrict__ x16,   // [B,T,I]
                                const _Float16* __restrict__ wih16, // [H,I]
                                const _Float16* __restrict__ whh16, // [H,H]
                                _Float16* __restrict__ hbuf0,       // [B,H]
                                _Float16* __restrict__ hbuf1,       // [B,H]
                                const float* __restrict__ bias_sum, // [H]
                                int* __restrict__ counter) {
  extern __shared__ _Float16 smem[];
  _Float16* Wl  = smem;                      // 32 x 640 f16 = 40 KB
  _Float16* xs0 = Wl + NCOL * KTOT;          // 64 x 128 f16 = 16 KB
  _Float16* xs1 = xs0 + Bz * Iz;             // 64 x 128 f16 = 16 KB

  const int tid  = threadIdx.x;
  const int wave = tid >> 5;        // 0..7
  const int lane = tid & 31;
  const int r    = lane & 15;
  const int half = lane >> 4;
  const int m0   = (wave >> 1) * 16;             // M tile base
  const int nt   = wave & 1;                     // N tile within WG slice
  const int n0g  = blockIdx.x * NCOL + nt * 16;  // global col base of tile

  // ---- stage weight slice into LDS (32 rows x 640 halves)
  const int wg_n0 = blockIdx.x * NCOL;
  for (int idx = tid; idx < NCOL * (KTOT / 8); idx += blockDim.x) {
    int row = idx / (KTOT / 8);
    int k   = (idx % (KTOT / 8)) * 8;
    int n   = wg_n0 + row;
    v8h v;
    if (k < Iz) v = *(const v8h*)(wih16 + (size_t)n * Iz + k);
    else        v = *(const v8h*)(whh16 + (size_t)n * Hz + (k - Iz));
    *(v8h*)(Wl + (size_t)row * KTOT + k) = v;
  }

  // ---- stage x_0, wait for weights + x staging
  stage_x(xs0, x16, 0, tid);
  wait_async();
  __syncthreads();

  const float bs   = bias_sum[n0g + r];  // per-column bias, constant over M
  const int   arow = m0 + r;             // A-matrix row handled by this lane
  const _Float16* wrow = Wl + (size_t)(nt * 16 + r) * KTOT;

  for (int t = 0; t < Tz; ++t) {
    const _Float16* __restrict__ hcur  = (t & 1) ? hbuf1 : hbuf0;
    _Float16* __restrict__       hnext = (t & 1) ? hbuf0 : hbuf1;
    const _Float16* xs_cur = (t & 1) ? xs1 : xs0;
    _Float16*       xs_nxt = (t & 1) ? xs0 : xs1;

    // kick off async staging of next timestep's x tile (overlaps WMMA work)
    if (t + 1 < Tz) stage_x(xs_nxt, x16, t + 1, tid);

    const _Float16* xsrow = xs_cur + (size_t)arow * Iz;
    const _Float16* hrow  = hcur + (size_t)arow * Hz;

    v8f c = {bs, bs, bs, bs, bs, bs, bs, bs};

    // x-part: K = 0..128 (A from LDS), fully unrolled -> 4 WMMA
#pragma unroll
    for (int k0 = 0; k0 < Iz; k0 += 32) {
      v16h a = a_frag(xsrow, k0, half);
      v16h b = *(const v16h*)(wrow + k0 + half * 16);
      c = __builtin_amdgcn_wmma_f32_16x16x32_f16(false, a, false, b,
                                                 (short)0, c, false, false);
    }
    // h-part: K = 0..512 (A from global, WGP$-cached), 8-deep unroll
#pragma unroll 8
    for (int k0 = 0; k0 < Hz; k0 += 32) {
      v16h a = a_frag(hrow, k0, half);
      v16h b = *(const v16h*)(wrow + Iz + k0 + half * 16);
      c = __builtin_amdgcn_wmma_f32_16x16x32_f16(false, a, false, b,
                                                 (short)0, c, false, false);
    }

    // branch-free tanh + scatter f16 h_new: c[v] -> (M=m0+v+8*half, N=n0g+r)
#pragma unroll
    for (int v = 0; v < 8; ++v) {
      float y = fast_tanh(c[v]);
      hnext[(size_t)(m0 + v + half * 8) * Hz + (n0g + r)] = (_Float16)y;
    }

    // make sure our async x-staging for t+1 landed before the step boundary
    wait_async();
    grid_barrier(counter, (t + 1) * NWG);
  }
}

// ---------------------------------------------------------------------------
// head: out[b] = sum_k h_last[b,k] * fc_w[k] + fc_b   (f32)
__global__ void head_kernel(const _Float16* __restrict__ hlast,
                            const float* __restrict__ fc_w,
                            const float* __restrict__ fc_b,
                            float* __restrict__ out) {
  __shared__ float red[256];
  int b = blockIdx.x;
  float s = 0.0f;
  for (int k = threadIdx.x; k < Hz; k += blockDim.x)
    s += (float)hlast[(size_t)b * Hz + k] * fc_w[k];
  red[threadIdx.x] = s;
  __syncthreads();
  for (int off = 128; off > 0; off >>= 1) {
    if ((int)threadIdx.x < off) red[threadIdx.x] += red[threadIdx.x + off];
    __syncthreads();
  }
  if (threadIdx.x == 0) out[b] = red[0] + fc_b[0];
}

// ---------------------------------------------------------------------------
extern "C" void kernel_launch(void* const* d_in, const int* in_sizes, int n_in,
                              void* d_out, int out_size, void* d_ws, size_t ws_size,
                              hipStream_t stream) {
  const float* x    = (const float*)d_in[0];  // [B,T,I]
  const float* W_ih = (const float*)d_in[1];  // [H,I]
  const float* W_hh = (const float*)d_in[2];  // [H,H]
  const float* b_ih = (const float*)d_in[3];  // [H]
  const float* b_hh = (const float*)d_in[4];  // [H]
  const float* fc_w = (const float*)d_in[5];  // [1,H]
  const float* fc_b = (const float*)d_in[6];  // [1]
  float* out = (float*)d_out;                 // [B,1]

  char* ws = (char*)d_ws;
  size_t off = 0;
  _Float16* x16   = (_Float16*)(ws + off); off += (size_t)Bz * Tz * Iz * sizeof(_Float16); // 16 MB
  _Float16* wih16 = (_Float16*)(ws + off); off += (size_t)Hz * Iz * sizeof(_Float16);      // 128 KB
  _Float16* whh16 = (_Float16*)(ws + off); off += (size_t)Hz * Hz * sizeof(_Float16);      // 512 KB
  _Float16* hbuf0 = (_Float16*)(ws + off); off += (size_t)Bz * Hz * sizeof(_Float16);      // 64 KB
  _Float16* hbuf1 = (_Float16*)(ws + off); off += (size_t)Bz * Hz * sizeof(_Float16);      // 64 KB
  float*    bias  = (float*)(ws + off);    off += (size_t)Hz * sizeof(float);              // 2 KB
  int*      cnt   = (int*)(ws + off);      off += 256;

  // 1) convert inputs/weights to f16 once per launch
  cvt_f32_to_f16<<<1024, 256, 0, stream>>>(x, x16, Bz * Tz * Iz);
  cvt_f32_to_f16<<<64,   256, 0, stream>>>(W_ih, wih16, Hz * Iz);
  cvt_f32_to_f16<<<256,  256, 0, stream>>>(W_hh, whh16, Hz * Hz);

  // 2) h0 = 0, fused bias, barrier counter = 0 (re-zeroed every launch)
  init_state<<<(Bz * Hz + 255) / 256, 256, 0, stream>>>(b_ih, b_hh, bias, hbuf0, cnt);

  // 3) persistent WMMA scan: 16 WGs x 256 threads, 72 KB dynamic LDS each
  size_t lds_bytes = (size_t)(NCOL * KTOT + 2 * Bz * Iz) * sizeof(_Float16);
  rnn_scan_kernel<<<NWG, 256, lds_bytes, stream>>>(
      x16, wih16, whh16, hbuf0, hbuf1, bias, cnt);

  // 4) final head; T is even so last h lands in hbuf0
  head_kernel<<<Bz, 256, 0, stream>>>(hbuf0, fc_w, fc_b, out);
}
